// AdvancedBrainStateClassifier_24086176596047
// MI455X (gfx1250) — matrix operand order repaired and checked by reading, not compiled, not run on hardware
//
#include <hip/hip_runtime.h>
#include <math.h>

// ---------------------------------------------------------------------------
// CDNA5 (gfx1250) implementation. wave32. Matrix work goes through
// v_wmma_f32_16x16x32_bf16; GEMM A-tiles are streamed into LDS with the
// Tensor Data Mover (tensor_load_to_lds + s_wait_tensorcnt, double-buffered,
// with D# LDS padding for bank-conflict-free fragment reads). All LDS operand
// tiles are stored K-contiguous so fragments load as ds_load_b128.
// ---------------------------------------------------------------------------

typedef __attribute__((ext_vector_type(16))) __bf16 v16bf;
typedef __attribute__((ext_vector_type(8)))  __bf16 v8bf;
typedef __attribute__((ext_vector_type(8)))  float  v8f;
typedef unsigned int u32x4 __attribute__((ext_vector_type(4)));
typedef int          i32x8 __attribute__((ext_vector_type(8)));
typedef int          i32x4 __attribute__((ext_vector_type(4)));

#define DEV __device__ __forceinline__

constexpr int N_NODES = 131072;
constexpr int N_EDGES = 1048576;
constexpr int T_SEQ   = 16;
constexpr int NHEADS  = 4;
constexpr int B_SEQ   = N_NODES / T_SEQ;   // 8192 sequences

#if defined(__has_builtin)
#  if __has_builtin(__builtin_amdgcn_tensor_load_to_lds)
#    define USE_TDM 1
#  endif
#  if __has_builtin(__builtin_amdgcn_s_wait_tensorcnt)
#    define HAVE_WAIT_TENSORCNT 1
#  endif
#endif
#ifndef USE_TDM
#  define USE_TDM 0
#endif
#ifndef HAVE_WAIT_TENSORCNT
#  define HAVE_WAIT_TENSORCNT 0
#endif

DEV __bf16 f2bf(float x) { return (__bf16)x; }

DEV v8f wmma_bf16(v16bf a, v16bf b, v8f c) {
  // (neg_a, A, neg_b, B, c_mod, C, reuse_a, reuse_b)
  return __builtin_amdgcn_wmma_f32_16x16x32_bf16(false, a, false, b, (short)0, c, false, false);
}

DEV v8bf lds8(const __bf16* p) { return *(const v8bf*)p; }   // 16B-aligned by construction
DEV v16bf cat16(v8bf lo, v8bf hi) {
  return __builtin_shufflevector(lo, hi, 0, 1, 2, 3, 4, 5, 6, 7, 8, 9, 10, 11, 12, 13, 14, 15);
}

// A-fragment (16-bit, 16x32, ISA 7.12.2): elements 0..7 hold K = 8*half+0..7,
// elements 8..15 hold K = 16+8*half+0..7 (relative to the 32-wide k step).
DEV v16bf load_a_frag(const __bf16* row, int k0, int half) {
  return cat16(lds8(row + k0 + 8 * half), lds8(row + k0 + 16 + 8 * half));
}
// B-fragment (32x16): elements 0..7 hold K = kbase + 8*half + 0..7,
// elements 8..15 hold K = kbase + 16 + 8*half + 0..7.
DEV v16bf load_b_frag(const __bf16* colrow, int kbase, int half) {
  const __bf16* p = colrow + kbase + 8 * half;
  return cat16(lds8(p), lds8(p + 16));
}

DEV float sigm(float x) { return 1.0f / (1.0f + __expf(-x)); }
// branch-free tanh via e^{2x}; clamp instead of libm's divergent range split
DEV float tanh_fast(float x) {
  float xc = fminf(fmaxf(x, -15.0f), 15.0f);
  float e  = __expf(2.0f * xc);
  return (e - 1.0f) / (e + 1.0f);
}

// monotonic float <-> uint key for atomicMax on floats of any sign
DEV unsigned encf(float f) { unsigned u = __float_as_uint(f); return (u & 0x80000000u) ? ~u : (u | 0x80000000u); }
DEV float    decf(unsigned u) { return __uint_as_float((u & 0x80000000u) ? (u ^ 0x80000000u) : ~u); }

#if USE_TDM
// TDM 2D tile load: rows x 32 bf16, row stride ld_elems, into LDS with
// 4-dword padding after every 16 dwords (row stride in LDS = 40 elements).
// D# per ISA 08 §8.3/8.4: pad_enable=1, pad_interval=3 (16 DW), pad_amount=3 (4 DW).
DEV void tdm_load_tile_bf16(const __bf16* gsrc, int ld_elems, void* ldsdst, int rows) {
  unsigned long long ga = (unsigned long long)(size_t)gsrc;
  unsigned lds = (unsigned)(size_t)ldsdst;
  u32x4 g0;
  g0.x = 1u;                                            // count=1 valid descriptor
  g0.y = lds;                                           // lds_addr (bytes)
  g0.z = (unsigned)ga;                                  // global_addr[31:0]
  g0.w = (unsigned)((ga >> 32) & 0x01ffffffu) | (2u << 30); // addr[56:32] | type=2
  unsigned d0 = (1u << 16)                              // data_size=1 -> 2 bytes
              | (1u << 20)                              // pad_enable
              | (3u << 22)                              // pad_interval: 16 DWORDs
              | (3u << 25);                             // pad_amount:   4 DWORDs
  unsigned td0 = 0x7fffffffu, td1 = 0x7fffffffu;        // huge tensor dims (no clip)
  unsigned d1 = (td0 & 0xffffu) << 16;
  unsigned d2 = ((td0 >> 16) & 0xffffu) | ((td1 & 0xffffu) << 16);
  unsigned d3 = ((td1 >> 16) & 0xffffu) | (32u << 16);  // tile_dim0 = 32 elems
  unsigned d4 = (unsigned)rows & 0xffffu;               // tile_dim1 = rows, tile_dim2=0
  unsigned d5 = (unsigned)ld_elems;                     // tensor_dim0_stride[31:0]
  unsigned d6 = 0u;
  unsigned d7 = 0u;
  i32x8 g1 = { (int)d0, (int)d1, (int)d2, (int)d3, (int)d4, (int)d5, (int)d6, (int)d7 };
  i32x4 z4 = { 0, 0, 0, 0 };
#if __clang_major__ >= 23
  i32x8 z8 = { 0, 0, 0, 0, 0, 0, 0, 0 };
  __builtin_amdgcn_tensor_load_to_lds(g0, g1, z4, z4, z8, 0);
#else
  __builtin_amdgcn_tensor_load_to_lds(g0, g1, z4, z4, 0);
#endif
}
#endif

DEV void wait_tensorcnt0() {
#if HAVE_WAIT_TENSORCNT
  __builtin_amdgcn_s_wait_tensorcnt(0);
#else
  asm volatile("s_wait_tensorcnt 0x0" ::: "memory");
#endif
}
DEV void wait_tensorcnt1() {
#if HAVE_WAIT_TENSORCNT
  __builtin_amdgcn_s_wait_tensorcnt(1);
#else
  asm volatile("s_wait_tensorcnt 0x1" ::: "memory");
#endif
}

// ---------------------------------------------------------------------------
// Generic GEMM: C(MxNcols) = bf16(A[MxK]) x bf16(B) (+bias0+bias1)
// TRANSB==0 : B = W, W row-major (K x Ncols)
// TRANSB==1 : B = W^T, W row-major (Ncols x K)   (PyTorch-style LSTM weights)
// block = 256 threads (8 waves), block tile = 64 rows x 32 cols, K <= 128.
// sA: TDM-fed padded tile [64][40] (rows stride 20 DW -> conflict-free b128).
// sB: K-contiguous [32][136] bf16 (stride 68 DW -> conflict-free b128).
// ---------------------------------------------------------------------------
template<int TRANSB>
__global__ void gemm_bf16_kernel(const __bf16* __restrict__ A, int lda, int K,
                                 const float* __restrict__ W, int Ncols,
                                 const float* __restrict__ bias0,
                                 const float* __restrict__ bias1,
                                 float* __restrict__ C, int ldc) {
  __shared__ __align__(16) __bf16 sA[2][64][40];   // TDM destination (padded rows)
  __shared__ __align__(16) __bf16 sB[32][136];     // sB[col][k]

  int tid  = threadIdx.x;
  int lane = tid & 31, wid = tid >> 5;
  int ln   = lane & 15, half = lane >> 4;
  int rowblock = blockIdx.x * 64;
  int rloc = (wid & 3) * 16;                 // row sub-tile within block
  int cloc = (wid >> 2) * 16;                // col sub-tile within block
  int cb   = blockIdx.y * 32;                // block col base
  int KT   = K / 32;

#if USE_TDM
  if (wid == 0) tdm_load_tile_bf16(A + (size_t)rowblock * lda, lda, &sA[0][0][0], 64);
#else
  for (int i = tid; i < 64 * 32; i += 256)
    sA[0][i >> 5][i & 31] = A[(size_t)(rowblock + (i >> 5)) * lda + (i & 31)];
#endif

  // stage B K-contiguous (convert f32 -> bf16 once)
  if (TRANSB) {
    int g = tid >> 3, sub = tid & 7;                 // 32 col groups x 8 threads
    for (int ks = sub * 8; ks < K; ks += 64) {
      const float* wp = W + (size_t)(cb + g) * K + ks;
      v8bf vv;
      #pragma unroll
      for (int e = 0; e < 8; e++) vv[e] = f2bf(wp[e]);
      *(v8bf*)&sB[g][ks] = vv;
    }
  } else {
    for (int i = tid; i < 32 * K; i += 256) {
      int c = i & 31, k = i >> 5;
      sB[c][k] = f2bf(W[(size_t)k * Ncols + cb + c]);
    }
  }

  v8f acc;
  #pragma unroll
  for (int i = 0; i < 8; i++) acc[i] = 0.0f;

  for (int kt = 0; kt < KT; ++kt) {
    int k0  = kt * 32;
    int cur = kt & 1, nxt = cur ^ 1;
#if USE_TDM
    if (wid == 0) {
      if (kt + 1 < KT) {
        tdm_load_tile_bf16(A + (size_t)rowblock * lda + (k0 + 32), lda, &sA[nxt][0][0], 64);
        wait_tensorcnt1();                   // tile kt done, kt+1 in flight
      } else {
        wait_tensorcnt0();
      }
    }
#else
    if (kt + 1 < KT)
      for (int i = tid; i < 64 * 32; i += 256)
        sA[nxt][i >> 5][i & 31] = A[(size_t)(rowblock + (i >> 5)) * lda + k0 + 32 + (i & 31)];
#endif
    __syncthreads();

    v16bf a = load_a_frag(&sA[cur][rloc + ln][0], 0, half);
    v16bf b = load_b_frag(&sB[cloc + ln][0], k0, half);
    acc = wmma_bf16(a, b, acc);
    __syncthreads();
  }

  int col = cb + cloc + ln;
  float bs = (bias0 ? bias0[col] : 0.0f) + (bias1 ? bias1[col] : 0.0f);
  #pragma unroll
  for (int r = 0; r < 8; r++) {
    int row = rowblock + rloc + r + 8 * half;    // C/D layout: vgpr r -> M = r + 8*half
    C[(size_t)row * ldc + col] = acc[r] + bs;
  }
}

// ---------------------------------------------------------------------------
// Frontend: x(3) -> 64 proj -> LN -> ELU -> 64x64 proj -> +PE -> bf16 seq0.
// one block (64 threads) per node.
// ---------------------------------------------------------------------------
__global__ void frontend_kernel(const float* __restrict__ x,
                                const float* fpW, const float* fpb,
                                const float* fpg, const float* fpbeta,
                                const float* ipW, const float* ipb,
                                __bf16* __restrict__ seq0) {
  int n = blockIdx.x, j = threadIdx.x;
  __shared__ float red[64];
  __shared__ float eh[64];

  float x0 = x[n * 3 + 0], x1 = x[n * 3 + 1], x2 = x[n * 3 + 2];
  float v = x0 * fpW[j] + x1 * fpW[64 + j] + x2 * fpW[128 + j] + fpb[j];

  red[j] = v; __syncthreads();
  for (int s = 32; s > 0; s >>= 1) { if (j < s) red[j] += red[j + s]; __syncthreads(); }
  float mean = red[0] * (1.0f / 64.0f); __syncthreads();
  float dv = v - mean;
  red[j] = dv * dv; __syncthreads();
  for (int s = 32; s > 0; s >>= 1) { if (j < s) red[j] += red[j + s]; __syncthreads(); }
  float var = red[0] * (1.0f / 64.0f); __syncthreads();

  float nm = dv * rsqrtf(var + 1e-5f) * fpg[j] + fpbeta[j];
  float e  = nm > 0.0f ? nm : (__expf(nm) - 1.0f);
  eh[j] = e; __syncthreads();

  float o = ipb[j];
  for (int k = 0; k < 64; k++) o += eh[k] * ipW[k * 64 + j];

  int   t  = n & (T_SEQ - 1);
  int   i2 = j >> 1;
  float d  = __expf(-(float)(2 * i2) * (9.210340371976184f / 64.0f)); // ln(10000)/H
  float pe = (j & 1) ? __cosf((float)t * d) : __sinf((float)t * d);

  seq0[(size_t)n * 64 + j] = f2bf(o + pe);
}

// ---------------------------------------------------------------------------
// LSTM recurrence (one direction). gates = x@Wih^T + bih + bhh precomputed.
// block = 128 threads (4 waves); each wave owns 16 sequences.
// sW[gatecol][k] = Whh[gatecol][k] in bf16 (K-contiguous, padded rows).
// Per step: D(16x256) = gates + h(16x64) x Whh^T via 32 WMMAs; the i/f/g/o
// nonlinearity needs no cross-lane traffic (tiles jt, jt+4, jt+8, jt+12
// share the lane/VGPR mapping).
// ---------------------------------------------------------------------------
__global__ void lstm_rec_kernel(const float* __restrict__ gates,
                                const float* __restrict__ Whh,   // (256 x 64) row-major
                                float* __restrict__ out,         // N x 128, write at colOffset
                                int colOffset, int reverse) {
  __shared__ __align__(16) __bf16 sW[256][72];     // 36 KB, stride 36 DW (conflict-free)
  __shared__ __align__(16) __bf16 shh[4][16][72];  // 9 KB per-wave h tile

  int tid = threadIdx.x;
  for (int col = tid; col < 256; col += 128) {
    const float* wp = Whh + (size_t)col * 64;
    for (int kc = 0; kc < 64; kc += 8) {
      v8bf vv;
      #pragma unroll
      for (int e = 0; e < 8; e++) vv[e] = f2bf(wp[kc + e]);
      *(v8bf*)&sW[col][kc] = vv;
    }
  }
  int wid = tid >> 5, lane = tid & 31, ln = lane & 15, half = lane >> 4;
  int seqbase = blockIdx.x * 64 + wid * 16;

  for (int idx = lane; idx < 16 * 72; idx += 32)
    shh[wid][idx / 72][idx % 72] = f2bf(0.0f);

  float cst[4][8];
  #pragma unroll
  for (int jt = 0; jt < 4; jt++)
    #pragma unroll
    for (int r = 0; r < 8; r++) cst[jt][r] = 0.0f;

  __syncthreads();

  for (int step = 0; step < T_SEQ; step++) {
    int t = reverse ? (T_SEQ - 1 - step) : step;

    v16bf a0 = load_a_frag(&shh[wid][ln][0], 0, half);    // h fragment K=0..31
    v16bf a1 = load_a_frag(&shh[wid][ln][0], 32, half);   // h fragment K=32..63

    #pragma unroll
    for (int jt = 0; jt < 4; jt++) {
      v8f g4[4];                                   // i, f, g, o tiles for cols jt*16..
      #pragma unroll
      for (int gs = 0; gs < 4; gs++) {
        int ct = gs * 4 + jt;
        v8f acc;
        #pragma unroll
        for (int r = 0; r < 8; r++) {
          int node = (seqbase + r + 8 * half) * T_SEQ + t;
          acc[r] = gates[(size_t)node * 256 + ct * 16 + ln];
        }
        v16bf b0 = load_b_frag(&sW[ct * 16 + ln][0], 0, half);
        v16bf b1 = load_b_frag(&sW[ct * 16 + ln][0], 32, half);
        acc = wmma_bf16(a0, b0, acc);
        acc = wmma_bf16(a1, b1, acc);
        g4[gs] = acc;
      }
      #pragma unroll
      for (int r = 0; r < 8; r++) {
        float cc = sigm(g4[1][r]) * cst[jt][r] + sigm(g4[0][r]) * tanh_fast(g4[2][r]);
        float hv = sigm(g4[3][r]) * tanh_fast(cc);
        cst[jt][r] = cc;
        int m = r + 8 * half;
        int node = (seqbase + m) * T_SEQ + t;
        shh[wid][m][jt * 16 + ln] = f2bf(hv);
        out[(size_t)node * 128 + colOffset + jt * 16 + ln] = hv;
      }
    }
    __syncthreads();
  }
}

// ---------------------------------------------------------------------------
// small utility kernels
// ---------------------------------------------------------------------------
__global__ void zero_u32_kernel(unsigned* __restrict__ p, size_t count) {
  size_t i = (size_t)blockIdx.x * blockDim.x + threadIdx.x;
  if (i < count) p[i] = 0u;
}

__global__ void f32_to_bf16_kernel(const float* __restrict__ in, __bf16* __restrict__ out, size_t count) {
  size_t i = (size_t)blockIdx.x * blockDim.x + threadIdx.x;
  if (i < count) out[i] = f2bf(in[i]);
}

// pack gat_W (4,128,16) into Wcat (128 x 64) with col = h*16+d
__global__ void gat_repack_kernel(const float* __restrict__ gatW, float* __restrict__ Wcat) {
  int i = blockIdx.x * blockDim.x + threadIdx.x;
  if (i < 128 * 64) {
    int k = i >> 6, c = i & 63, h = c >> 4, d = c & 15;
    Wcat[i] = gatW[h * (128 * 16) + k * 16 + d];
  }
}

__global__ void sdot_kernel(const float* __restrict__ hh,
                            const float* __restrict__ asrc, const float* __restrict__ adst,
                            float* __restrict__ ssrc, float* __restrict__ sdst) {
  int i = blockIdx.x * blockDim.x + threadIdx.x;
  if (i >= N_NODES * NHEADS) return;
  int n = i >> 2, h = i & 3;
  const float* hp = hh + (size_t)n * 64 + h * 16;
  float s1 = 0.0f, s2 = 0.0f;
  for (int d = 0; d < 16; d++) { s1 += hp[d] * asrc[h * 16 + d]; s2 += hp[d] * adst[h * 16 + d]; }
  ssrc[i] = s1; sdst[i] = s2;
}

__global__ void edge_max_kernel(const int* __restrict__ ei,
                                const float* __restrict__ ssrc, const float* __restrict__ sdst,
                                unsigned* __restrict__ mkey) {
  int gid = blockIdx.x * blockDim.x + threadIdx.x;
  if (gid >= (N_EDGES + N_NODES) * NHEADS) return;
  int e = gid >> 2, h = gid & 3;
  int s, d;
  if (e < N_EDGES) { s = ei[e]; d = ei[N_EDGES + e]; } else { s = d = e - N_EDGES; }
  float v = ssrc[s * 4 + h] + sdst[d * 4 + h];
  v = v > 0.0f ? v : 0.2f * v;                       // leaky_relu(0.2)
  atomicMax(&mkey[d * 4 + h], encf(v));
}

__global__ void edge_acc_kernel(const int* __restrict__ ei,
                                const float* __restrict__ ssrc, const float* __restrict__ sdst,
                                const unsigned* __restrict__ mkey,
                                const float* __restrict__ hh,
                                float* __restrict__ den, float* __restrict__ gatacc) {
  int gid = blockIdx.x * blockDim.x + threadIdx.x;
  if (gid >= (N_EDGES + N_NODES) * NHEADS) return;
  int e = gid >> 2, h = gid & 3;
  int s, d;
  if (e < N_EDGES) { s = ei[e]; d = ei[N_EDGES + e]; } else { s = d = e - N_EDGES; }
  float v = ssrc[s * 4 + h] + sdst[d * 4 + h];
  v = v > 0.0f ? v : 0.2f * v;
  float w = __expf(v - decf(mkey[d * 4 + h]));
  atomicAdd(&den[d * 4 + h], w);
  const float* hp = hh + (size_t)s * 64 + h * 16;
  float* oa = gatacc + (size_t)d * 64 + h * 16;
  #pragma unroll
  for (int k = 0; k < 16; k++) atomicAdd(&oa[k], w * hp[k]);
}

__global__ void gat_fin_kernel(const float* __restrict__ gatacc, const float* __restrict__ den,
                               const float* __restrict__ gbias,
                               float* __restrict__ gat_out, __bf16* __restrict__ gat_out_bf) {
  int i = blockIdx.x * blockDim.x + threadIdx.x;
  if (i >= N_NODES * 64) return;
  int n = i >> 6, c = i & 63, h = c >> 4;
  float v = gatacc[i] / den[n * 4 + h] + gbias[c];
  gat_out[i] = v;
  gat_out_bf[i] = f2bf(v);
}

__global__ void ln64_kernel(const float* __restrict__ in,
                            const float* g, const float* beta,
                            float* __restrict__ out) {
  int n = blockIdx.x, j = threadIdx.x;
  __shared__ float red[64];
  float v = in[(size_t)n * 64 + j];
  red[j] = v; __syncthreads();
  for (int s = 32; s > 0; s >>= 1) { if (j < s) red[j] += red[j + s]; __syncthreads(); }
  float mean = red[0] * (1.0f / 64.0f); __syncthreads();
  float dv = v - mean;
  red[j] = dv * dv; __syncthreads();
  for (int s = 32; s > 0; s >>= 1) { if (j < s) red[j] += red[j + s]; __syncthreads(); }
  float var = red[0] * (1.0f / 64.0f); __syncthreads();
  out[(size_t)n * 64 + j] = dv * rsqrtf(var + 1e-5f) * g[j] + beta[j];
}

__global__ void pool_kernel(const float* __restrict__ g_out,
                            float* __restrict__ psum, unsigned* __restrict__ pmax) {
  int tid = threadIdx.x;
  int c = tid & 63;
  int n0 = blockIdx.x * 4 + (tid >> 6);
  float s = 0.0f, mx = -3.4e38f;
  for (int n = n0; n < N_NODES; n += gridDim.x * 4) {
    float v = g_out[(size_t)n * 64 + c];
    s += v; mx = fmaxf(mx, v);
  }
  atomicAdd(&psum[c], s);
  atomicMax(&pmax[c], encf(mx));
}

// LN over first L (<=128, pow2) entries of v + optional ELU; 192-thread block.
DEV void ln_elu_shared(float* v, int L, const float* g, const float* be,
                       float* red, int t, bool do_elu) {
  red[t] = (t < L) ? v[t] : 0.0f; __syncthreads();
  if (t < 64) red[t] += red[t + 64] + red[t + 128];
  __syncthreads();
  for (int s = 32; s > 0; s >>= 1) { if (t < s) red[t] += red[t + s]; __syncthreads(); }
  float mean = red[0] / (float)L; __syncthreads();
  float dv = (t < L) ? (v[t] - mean) : 0.0f;
  red[t] = dv * dv; __syncthreads();
  if (t < 64) red[t] += red[t + 64] + red[t + 128];
  __syncthreads();
  for (int s = 32; s > 0; s >>= 1) { if (t < s) red[t] += red[t + s]; __syncthreads(); }
  float var = red[0] / (float)L; __syncthreads();
  if (t < L) {
    float nm = dv * rsqrtf(var + 1e-5f) * g[t] + be[t];
    v[t] = do_elu ? (nm > 0.0f ? nm : (__expf(nm) - 1.0f)) : nm;
  }
  __syncthreads();
}

__global__ void tail_kernel(const float* __restrict__ psum, const unsigned* __restrict__ pmax,
                            const float* fu1W, const float* fu1b, const float* fu1g, const float* fu1be,
                            const float* fu2W, const float* fu2b, const float* fu2g, const float* fu2be,
                            const float* c1W, const float* c1b, const float* c1g, const float* c1be,
                            const float* c2W, const float* c2b, const float* c2g, const float* c2be,
                            const float* c3W, const float* c3b,
                            float* __restrict__ out) {
  __shared__ float v0[192], v1[128], red[192];
  int t = threadIdx.x;
  if (t < 64) {
    v0[t]       = decf(pmax[t]);
    v0[64 + t]  = psum[t] * (1.0f / (float)N_NODES);
    v0[128 + t] = psum[t];
  }
  __syncthreads();
  if (t < 128) { float a = fu1b[t]; for (int k = 0; k < 192; k++) a += v0[k] * fu1W[k * 128 + t]; v1[t] = a; }
  __syncthreads();
  ln_elu_shared(v1, 128, fu1g, fu1be, red, t, true);
  if (t < 64) { float a = fu2b[t]; for (int k = 0; k < 128; k++) a += v1[k] * fu2W[k * 64 + t]; v0[t] = a; }
  __syncthreads();
  ln_elu_shared(v0, 64, fu2g, fu2be, red, t, true);
  if (t < 32) { float a = c1b[t]; for (int k = 0; k < 64; k++) a += v0[k] * c1W[k * 32 + t]; v1[t] = a; }
  __syncthreads();
  ln_elu_shared(v1, 32, c1g, c1be, red, t, true);
  if (t < 16) { float a = c2b[t]; for (int k = 0; k < 32; k++) a += v1[k] * c2W[k * 16 + t]; v0[t] = a; }
  __syncthreads();
  ln_elu_shared(v0, 16, c2g, c2be, red, t, true);
  if (t < 6) { float a = c3b[t]; for (int k = 0; k < 16; k++) a += v0[k] * c3W[k * 6 + t]; out[t] = a; }
}

// ---------------------------------------------------------------------------
// host launcher
// ---------------------------------------------------------------------------
extern "C" void kernel_launch(void* const* d_in, const int* in_sizes, int n_in,
                              void* d_out, int out_size, void* d_ws, size_t ws_size,
                              hipStream_t stream) {
  (void)in_sizes; (void)n_in; (void)out_size; (void)ws_size;
  const float* x   = (const float*)d_in[0];
  const int*   ei  = (const int*)d_in[1];
  const float* fpW = (const float*)d_in[2];
  const float* fpb = (const float*)d_in[3];
  const float* fpg = (const float*)d_in[4];
  const float* fpbe= (const float*)d_in[5];
  const float* ipW = (const float*)d_in[6];
  const float* ipb = (const float*)d_in[7];
  // lstm params: [8..23] = l0f(Wih,Whh,bih,bhh), l0b(...), l1f(...), l1b(...)
  const float* lW[4][4];
  for (int di = 0; di < 4; di++)
    for (int pi = 0; pi < 4; pi++)
      lW[di][pi] = (const float*)d_in[8 + di * 4 + pi];
  const float* gatW   = (const float*)d_in[24];
  const float* gatAs  = (const float*)d_in[25];
  const float* gatAd  = (const float*)d_in[26];
  const float* gatB   = (const float*)d_in[27];
  const float* opW    = (const float*)d_in[28];
  const float* opB    = (const float*)d_in[29];
  const float* mhaG   = (const float*)d_in[30];
  const float* mhaBe  = (const float*)d_in[31];
  const float* fu1W   = (const float*)d_in[32];
  const float* fu1b   = (const float*)d_in[33];
  const float* fu1g   = (const float*)d_in[34];
  const float* fu1be  = (const float*)d_in[35];
  const float* fu2W   = (const float*)d_in[36];
  const float* fu2b   = (const float*)d_in[37];
  const float* fu2g   = (const float*)d_in[38];
  const float* fu2be  = (const float*)d_in[39];
  const float* c1W    = (const float*)d_in[40];
  const float* c1b    = (const float*)d_in[41];
  const float* c1g    = (const float*)d_in[42];
  const float* c1be   = (const float*)d_in[43];
  const float* c2W    = (const float*)d_in[44];
  const float* c2b    = (const float*)d_in[45];
  const float* c2g    = (const float*)d_in[46];
  const float* c2be   = (const float*)d_in[47];
  const float* c3W    = (const float*)d_in[48];
  const float* c3b    = (const float*)d_in[49];
  float* out = (float*)d_out;

  // workspace layout (zeroed region first; all sizes 256B-aligned)
  char* w = (char*)d_ws;
  auto take = [&](size_t bytes) { void* p = (void*)w; w += (bytes + 255) & ~(size_t)255; return p; };

  unsigned* mkey   = (unsigned*)take((size_t)N_NODES * 4 * 4);
  float*    den    = (float*)take((size_t)N_NODES * 4 * 4);
  float*    gatacc = (float*)take((size_t)N_NODES * 64 * 4);
  float*    psum   = (float*)take(64 * 4);
  unsigned* pmax   = (unsigned*)take(64 * 4);
  size_t zero_words = (size_t)N_NODES * 4 + (size_t)N_NODES * 4 + (size_t)N_NODES * 64 + 64 + 64;

  __bf16* seq0    = (__bf16*)take((size_t)N_NODES * 64 * 2);
  float*  gates   = (float*)take((size_t)N_NODES * 256 * 4);
  float*  dir1    = (float*)take((size_t)N_NODES * 128 * 4);
  __bf16* seq1    = (__bf16*)take((size_t)N_NODES * 128 * 2);
  float*  dir2    = (float*)take((size_t)N_NODES * 128 * 4);
  __bf16* h2bf    = (__bf16*)take((size_t)N_NODES * 128 * 2);
  float*  Wcat    = (float*)take(128 * 64 * 4);
  float*  hh      = (float*)take((size_t)N_NODES * 64 * 4);
  float*  ssrc    = (float*)take((size_t)N_NODES * 4 * 4);
  float*  sdst    = (float*)take((size_t)N_NODES * 4 * 4);
  float*  gatout  = (float*)take((size_t)N_NODES * 64 * 4);
  __bf16* gatoutb = (__bf16*)take((size_t)N_NODES * 64 * 2);
  float*  optmp   = (float*)take((size_t)N_NODES * 64 * 4);
  float*  gout    = (float*)take((size_t)N_NODES * 64 * 4);

  // 0) zero accumulators
  zero_u32_kernel<<<(unsigned)((zero_words + 1023) / 1024), 1024, 0, stream>>>(mkey, zero_words);

  // 1) frontend
  frontend_kernel<<<N_NODES, 64, 0, stream>>>(x, fpW, fpb, fpg, fpbe, ipW, ipb, seq0);

  dim3 g256(N_NODES / 64, 256 / 32);
  dim3 g64 (N_NODES / 64, 64 / 32);
  dim3 recg(B_SEQ / 64);

  // 2) LSTM layer 0 (din=64)
  gemm_bf16_kernel<1><<<g256, 256, 0, stream>>>(seq0, 64, 64, lW[0][0], 256, lW[0][2], lW[0][3], gates, 256);
  lstm_rec_kernel<<<recg, 128, 0, stream>>>(gates, lW[0][1], dir1, 0, 0);
  gemm_bf16_kernel<1><<<g256, 256, 0, stream>>>(seq0, 64, 64, lW[1][0], 256, lW[1][2], lW[1][3], gates, 256);
  lstm_rec_kernel<<<recg, 128, 0, stream>>>(gates, lW[1][1], dir1, 64, 1);
  f32_to_bf16_kernel<<<(unsigned)(((size_t)N_NODES * 128 + 255) / 256), 256, 0, stream>>>(dir1, seq1, (size_t)N_NODES * 128);

  // 3) LSTM layer 1 (din=128)
  gemm_bf16_kernel<1><<<g256, 256, 0, stream>>>(seq1, 128, 128, lW[2][0], 256, lW[2][2], lW[2][3], gates, 256);
  lstm_rec_kernel<<<recg, 128, 0, stream>>>(gates, lW[2][1], dir2, 0, 0);
  gemm_bf16_kernel<1><<<g256, 256, 0, stream>>>(seq1, 128, 128, lW[3][0], 256, lW[3][2], lW[3][3], gates, 256);
  lstm_rec_kernel<<<recg, 128, 0, stream>>>(gates, lW[3][1], dir2, 64, 1);
  f32_to_bf16_kernel<<<(unsigned)(((size_t)N_NODES * 128 + 255) / 256), 256, 0, stream>>>(dir2, h2bf, (size_t)N_NODES * 128);

  // 4) GAT
  gat_repack_kernel<<<(128 * 64 + 255) / 256, 256, 0, stream>>>(gatW, Wcat);
  gemm_bf16_kernel<0><<<g64, 256, 0, stream>>>(h2bf, 128, 128, Wcat, 64, nullptr, nullptr, hh, 64);
  sdot_kernel<<<(N_NODES * 4 + 255) / 256, 256, 0, stream>>>(hh, gatAs, gatAd, ssrc, sdst);
  int etot = (N_EDGES + N_NODES) * NHEADS;
  edge_max_kernel<<<(etot + 255) / 256, 256, 0, stream>>>(ei, ssrc, sdst, mkey);
  edge_acc_kernel<<<(etot + 255) / 256, 256, 0, stream>>>(ei, ssrc, sdst, mkey, hh, den, gatacc);
  gat_fin_kernel<<<(N_NODES * 64 + 255) / 256, 256, 0, stream>>>(gatacc, den, gatB, gatout, gatoutb);

  // 5) output projection + LN
  gemm_bf16_kernel<0><<<g64, 256, 0, stream>>>(gatoutb, 64, 64, opW, 64, opB, nullptr, optmp, 64);
  ln64_kernel<<<N_NODES, 64, 0, stream>>>(optmp, mhaG, mhaBe, gout);

  // 6) pooling + tail MLP
  pool_kernel<<<512, 256, 0, stream>>>(gout, psum, pmax);
  tail_kernel<<<1, 192, 0, stream>>>(psum, pmax,
                                     fu1W, fu1b, fu1g, fu1be,
                                     fu2W, fu2b, fu2g, fu2be,
                                     c1W, c1b, c1g, c1be,
                                     c2W, c2b, c2g, c2be,
                                     c3W, c3b, out);
}
